// SoluModel_11201274708450
// MI455X (gfx1250) — matrix operand order repaired and checked by reading, not compiled, not run on hardware
//
#include <hip/hip_runtime.h>
#include <hip/hip_bf16.h>
#include <math.h>

// ---------------------------------------------------------------------------
// CDNA5 WMMA (gfx1250, wave32) helpers
// ---------------------------------------------------------------------------
#define NT   256     // threads per block (8 wave32 waves)
#define BM   128     // M tile (8 waves x 16 rows)
#define BN   64      // N tile (4 x 16 col groups per wave)
#define BK   32      // K tile (16x16x32 bf16 WMMA)
#define KP   (BK/2)  // 16 packed u32 K-pairs per row
#define LDP  (KP+1)  // padded LDS row stride in u32 (17: conflict-free)

typedef __attribute__((ext_vector_type(16))) __bf16       bf16x16;
typedef __attribute__((ext_vector_type(8)))  float        floatx8;
typedef __attribute__((ext_vector_type(8)))  unsigned int uintx8;

union fragU { uintx8 u; bf16x16 b; };

__device__ __forceinline__ unsigned short f2bfu(float f) {
  union { float f; unsigned int u; } x; x.f = f;
  unsigned int r = 0x7FFFu + ((x.u >> 16) & 1u);   // round-to-nearest-even
  return (unsigned short)((x.u + r) >> 16);
}
__device__ __forceinline__ unsigned int pack2(float lo, float hi) {
  return (unsigned int)f2bfu(lo) | ((unsigned int)f2bfu(hi) << 16);
}

// A fragment (16x32 bf16) from LDS packed as [m][kpair] u32.
// Contiguous K order (matches frag_b32): VGPR v holds K=16*half+2v,2v+1
// ->  u32 pair index: kp = 8*half+v.
__device__ __forceinline__ bf16x16 frag_a32(const unsigned int* s, int mOff,
                                            int lane) {
  const int half = lane >> 4;
  const unsigned int* row = s + (mOff + (lane & 15)) * LDP;
  fragU f;
#pragma unroll
  for (int v = 0; v < 4; ++v) f.u[v] = row[8 * half + v];
#pragma unroll
  for (int v = 0; v < 4; ++v) f.u[4 + v] = row[8 * half + 4 + v];
  return f.b;
}
// B fragment (32x16 bf16) from LDS packed as [n][kpair] u32.
// Lanes 0-15: N=lane, K=0..15 ; lanes 16-31: K=16..31. VGPR v: kp = 8*half+v.
__device__ __forceinline__ bf16x16 frag_b32(const unsigned int* s, int nOff,
                                            int lane) {
  const int half = lane >> 4;
  const unsigned int* row = s + (nOff + (lane & 15)) * LDP;
  fragU f;
#pragma unroll
  for (int v = 0; v < 8; ++v) f.u[v] = row[8 * half + v];
  return f.b;
}
__device__ __forceinline__ floatx8 wmma_bf16(bf16x16 a, bf16x16 b, floatx8 c) {
  return __builtin_amdgcn_wmma_f32_16x16x32_bf16(false, a, false, b,
                                                 (short)0, c, false, false);
}

// ---------------------------------------------------------------------------
// Conv1d (kernel=3, pad=1) as implicit GEMM, bf16 WMMA:
//   Y[b,co,l] = bias[co] + sum_{tap,ci} W[co,ci,tap] * X[b,ci,l+tap-1]
// ---------------------------------------------------------------------------
__global__ __launch_bounds__(NT) void conv1d_wmma_k(
    const float* __restrict__ X, const float* __restrict__ W,
    const float* __restrict__ bias, float* __restrict__ Y,
    int Cin, int Cout, int L) {
  __shared__ unsigned int sA[BM * LDP];  // [m][kp]
  __shared__ unsigned int sB[BN * LDP];  // [n][kp]
  const int b = blockIdx.z;
  const int mBase = blockIdx.y * BM;
  const int nBase = blockIdx.x * BN;
  const int tid = threadIdx.x;
  const int lane = tid & 31, wid = tid >> 5;
  const int mw = wid * 16;  // wave M strip
  const float* Xb = X + (size_t)b * Cin * L;
  const bool aFull = (mBase + BM <= Cout);
  const bool bFull = (nBase >= 1) && (nBase + BN + 1 <= L);

  floatx8 c0, c1, c2, c3;
  {
    const int half = lane >> 4;
#pragma unroll
    for (int r = 0; r < 8; ++r) {
      int row = mBase + mw + r + 8 * half;
      float bv = (bias != nullptr && row < Cout) ? bias[row] : 0.f;
      c0[r] = bv; c1[r] = bv; c2[r] = bv; c3[r] = bv;
    }
  }

  for (int kb = 0; kb < Cin; kb += BK) {
    if (kb + BK < Cin)  // next weight K-slab -> global_prefetch_b8
      __builtin_prefetch(&W[((size_t)(mBase + (tid >> 1)) * Cin + kb + BK) * 3], 0, 3);
    for (int tap = 0; tap < 3; ++tap) {
      __syncthreads();
      // ---- A tile: 128 rows x 16 k-pairs (8 pairs/thread) ----
      if (aFull) {
#pragma unroll
        for (int t = 0; t < (BM * KP) / NT; ++t) {
          int e = tid + t * NT;
          int i = e >> 4, jp = e & 15;
          const float* wp = &W[((size_t)(mBase + i) * Cin + kb + 2 * jp) * 3 + tap];
          sA[i * LDP + jp] = pack2(wp[0], wp[3]);
        }
      } else {
#pragma unroll
        for (int t = 0; t < (BM * KP) / NT; ++t) {
          int e = tid + t * NT;
          int i = e >> 4, jp = e & 15;
          int co = mBase + i;
          float lo = 0.f, hi = 0.f;
          if (co < Cout) {
            const float* wp = &W[((size_t)co * Cin + kb + 2 * jp) * 3 + tap];
            lo = wp[0]; hi = wp[3];
          }
          sA[i * LDP + jp] = pack2(lo, hi);
        }
      }
      // ---- B tile (transposed to [n][kp]): 16 k-pairs x 64 cols ----
      if (bFull) {
#pragma unroll
        for (int t = 0; t < (KP * BN) / NT; ++t) {
          int e = tid + t * NT;
          int kp = e >> 6, n = e & 63;
          const float* xp = &Xb[(size_t)(kb + 2 * kp) * L + nBase + n + tap - 1];
          sB[n * LDP + kp] = pack2(xp[0], xp[L]);
        }
      } else {
#pragma unroll
        for (int t = 0; t < (KP * BN) / NT; ++t) {
          int e = tid + t * NT;
          int kp = e >> 6, n = e & 63;
          int l = nBase + n + tap - 1;
          float lo = 0.f, hi = 0.f;
          if (l >= 0 && l < L) {
            const float* xp = &Xb[(size_t)(kb + 2 * kp) * L + l];
            lo = xp[0]; hi = xp[L];
          }
          sB[n * LDP + kp] = pack2(lo, hi);
        }
      }
      __syncthreads();
      bf16x16 a  = frag_a32(sA, mw, lane);
      bf16x16 b0 = frag_b32(sB, 0, lane);
      bf16x16 b1 = frag_b32(sB, 16, lane);
      bf16x16 b2 = frag_b32(sB, 32, lane);
      bf16x16 b3 = frag_b32(sB, 48, lane);
      c0 = wmma_bf16(a, b0, c0);
      c1 = wmma_bf16(a, b1, c1);
      c2 = wmma_bf16(a, b2, c2);
      c3 = wmma_bf16(a, b3, c3);
    }
  }

  const int half = lane >> 4, mr = lane & 15;
#pragma unroll
  for (int r = 0; r < 8; ++r) {
    int row = mBase + mw + r + 8 * half;
    if (row < Cout) {
      float* yr = &Y[((size_t)b * Cout + row) * L];
      int col = nBase + mr;
      if (col < L)      yr[col]      = c0[r];
      if (col + 16 < L) yr[col + 16] = c1[r];
      if (col + 32 < L) yr[col + 32] = c2[r];
      if (col + 48 < L) yr[col + 48] = c3[r];
    }
  }
}

// ---------------------------------------------------------------------------
// Generic batched GEMM (bf16 WMMA): C = A*B (+bias[n]) (+sigmoid)
// A:[M,K](lda), B:[K,N](ldb) or transB: B[n*ldb+k], C:[M,N](ldc)
// ---------------------------------------------------------------------------
__global__ __launch_bounds__(NT) void gemm_wmma_k(
    const float* __restrict__ A, const float* __restrict__ Bm,
    const float* __restrict__ bias, float* __restrict__ C,
    int M, int N, int K,
    long long sAb, long long sBb, long long sCb,
    int lda, int ldb, int ldc, int transB, int act) {
  __shared__ unsigned int sA[BM * LDP];
  __shared__ unsigned int sB[BN * LDP];
  const int bz = blockIdx.z;
  const float* Ab = A  + (size_t)bz * sAb;
  const float* Bb = Bm + (size_t)bz * sBb;
  float*       Cb = C  + (size_t)bz * sCb;
  const int mBase = blockIdx.y * BM;
  const int nBase = blockIdx.x * BN;
  const int tid = threadIdx.x;
  const int lane = tid & 31, wid = tid >> 5;
  const int mw = wid * 16;
  const int half = lane >> 4, mr = lane & 15;

  floatx8 c0, c1, c2, c3;
  {
    int col = nBase + mr;
    float v0 = (bias && col < N)      ? bias[col]      : 0.f;
    float v1 = (bias && col + 16 < N) ? bias[col + 16] : 0.f;
    float v2 = (bias && col + 32 < N) ? bias[col + 32] : 0.f;
    float v3 = (bias && col + 48 < N) ? bias[col + 48] : 0.f;
#pragma unroll
    for (int r = 0; r < 8; ++r) { c0[r] = v0; c1[r] = v1; c2[r] = v2; c3[r] = v3; }
  }

  for (int kb = 0; kb < K; kb += BK) {
    __syncthreads();
    const bool kFull = (kb + BK <= K);
    // ---- A tile ----
    if (kFull && (mBase + BM <= M)) {
#pragma unroll
      for (int t = 0; t < (BM * KP) / NT; ++t) {
        int e = tid + t * NT;
        int i = e >> 4, jp = e & 15;
        const float* ap = &Ab[(size_t)(mBase + i) * lda + kb + 2 * jp];
        sA[i * LDP + jp] = pack2(ap[0], ap[1]);
      }
    } else {
#pragma unroll
      for (int t = 0; t < (BM * KP) / NT; ++t) {
        int e = tid + t * NT;
        int i = e >> 4, jp = e & 15;
        int m = mBase + i, k = kb + 2 * jp;
        float lo = (m < M && k < K)     ? Ab[(size_t)m * lda + k]     : 0.f;
        float hi = (m < M && k + 1 < K) ? Ab[(size_t)m * lda + k + 1] : 0.f;
        sA[i * LDP + jp] = pack2(lo, hi);
      }
    }
    // ---- B tile ([n][kp]) ----
    if (kFull && (nBase + BN <= N)) {
#pragma unroll
      for (int t = 0; t < (KP * BN) / NT; ++t) {
        int e = tid + t * NT;
        int kp = e >> 6, n = e & 63;
        if (transB) {
          const float* bp = &Bb[(size_t)(nBase + n) * ldb + kb + 2 * kp];
          sB[n * LDP + kp] = pack2(bp[0], bp[1]);
        } else {
          const float* bp = &Bb[(size_t)(kb + 2 * kp) * ldb + nBase + n];
          sB[n * LDP + kp] = pack2(bp[0], bp[ldb]);
        }
      }
    } else {
#pragma unroll
      for (int t = 0; t < (KP * BN) / NT; ++t) {
        int e = tid + t * NT;
        int kp = e >> 6, n = e & 63;
        int nn = nBase + n, k = kb + 2 * kp;
        float lo = 0.f, hi = 0.f;
        if (nn < N) {
          if (transB) {
            if (k < K)     lo = Bb[(size_t)nn * ldb + k];
            if (k + 1 < K) hi = Bb[(size_t)nn * ldb + k + 1];
          } else {
            if (k < K)     lo = Bb[(size_t)k * ldb + nn];
            if (k + 1 < K) hi = Bb[(size_t)(k + 1) * ldb + nn];
          }
        }
        sB[n * LDP + kp] = pack2(lo, hi);
      }
    }
    __syncthreads();
    bf16x16 a  = frag_a32(sA, mw, lane);
    bf16x16 b0 = frag_b32(sB, 0, lane);
    bf16x16 b1 = frag_b32(sB, 16, lane);
    bf16x16 b2 = frag_b32(sB, 32, lane);
    bf16x16 b3 = frag_b32(sB, 48, lane);
    c0 = wmma_bf16(a, b0, c0);
    c1 = wmma_bf16(a, b1, c1);
    c2 = wmma_bf16(a, b2, c2);
    c3 = wmma_bf16(a, b3, c3);
  }

#pragma unroll
  for (int r = 0; r < 8; ++r) {
    int row = mBase + mw + r + 8 * half;
    if (row < M) {
      float* cr = &Cb[(size_t)row * ldc];
      int col = nBase + mr;
      float v;
      if (col < N)      { v = c0[r]; if (act) v = 1.f / (1.f + expf(-v)); cr[col]      = v; }
      if (col + 16 < N) { v = c1[r]; if (act) v = 1.f / (1.f + expf(-v)); cr[col + 16] = v; }
      if (col + 32 < N) { v = c2[r]; if (act) v = 1.f / (1.f + expf(-v)); cr[col + 32] = v; }
      if (col + 48 < N) { v = c3[r]; if (act) v = 1.f / (1.f + expf(-v)); cr[col + 48] = v; }
    }
  }
}

// ---------------------------------------------------------------------------
// LayerNorm over (C,L) per batch, in place. grid = B blocks.
// ---------------------------------------------------------------------------
__global__ __launch_bounds__(NT) void layernorm2_k(
    float* __restrict__ X, const float* __restrict__ g,
    const float* __restrict__ be, int C, int L) {
  __shared__ float red[NT];
  __shared__ float s_mu, s_rstd;
  const int b = blockIdx.x, tid = threadIdx.x;
  const size_t n = (size_t)C * L;
  float* Xb = X + (size_t)b * n;
  float sum = 0.f, sq = 0.f;
  for (size_t i = tid; i < n; i += NT) { float v = Xb[i]; sum += v; sq += v * v; }
  red[tid] = sum; __syncthreads();
  for (int s = NT / 2; s > 0; s >>= 1) { if (tid < s) red[tid] += red[tid + s]; __syncthreads(); }
  float tot = red[0]; __syncthreads();
  red[tid] = sq; __syncthreads();
  for (int s = NT / 2; s > 0; s >>= 1) { if (tid < s) red[tid] += red[tid + s]; __syncthreads(); }
  if (tid == 0) {
    float mu = tot / (float)n;
    float var = red[0] / (float)n - mu * mu;
    s_mu = mu; s_rstd = rsqrtf(var + 1e-5f);
  }
  __syncthreads();
  const float mu = s_mu, rs = s_rstd;
  for (size_t i = tid; i < n; i += NT)
    Xb[i] = (Xb[i] - mu) * rs * g[i] + be[i];
}

// ---------------------------------------------------------------------------
// Softmax over L for sw[b,ch,:]. grid = (2, B). L <= 224 < NT.
// ---------------------------------------------------------------------------
__global__ __launch_bounds__(NT) void softmax_k(
    const float* __restrict__ sw, float* __restrict__ sout,
    float* __restrict__ wout, int L) {
  __shared__ float red[NT];
  const int ch = blockIdx.x, b = blockIdx.y, tid = threadIdx.x;
  const float* x = sw + ((size_t)b * 2 + ch) * L;
  float* o = (ch == 0 ? sout : wout) + (size_t)b * L;
  float v = (tid < L) ? x[tid] : -3.0e38f;
  red[tid] = v; __syncthreads();
  for (int s = NT / 2; s > 0; s >>= 1) { if (tid < s) red[tid] = fmaxf(red[tid], red[tid + s]); __syncthreads(); }
  const float mx = red[0]; __syncthreads();
  float e = (tid < L) ? expf(v - mx) : 0.f;
  red[tid] = e; __syncthreads();
  for (int s = NT / 2; s > 0; s >>= 1) { if (tid < s) red[tid] += red[tid + s]; __syncthreads(); }
  if (tid < L) o[tid] = e / red[0];
}

// ---------------------------------------------------------------------------
// Gaussian pooling weights: Wg[b,i,j] = (G/rowmax(G)) * w[b,j]. grid=(L,B).
// ---------------------------------------------------------------------------
__global__ __launch_bounds__(NT) void gaussw_k(
    const float* __restrict__ sbuf, const float* __restrict__ wbuf,
    float* __restrict__ Wg, int L) {
  __shared__ float red[NT];
  const int i = blockIdx.x, b = blockIdx.y, j = threadIdx.x;
  const float stdv = 0.1f * (float)L * sbuf[(size_t)b * L + i];
  const float inv = 1.f / (1e-5f + 2.f * stdv * stdv);
  const float coef = 0.3989422804f / stdv;  // (2*pi)^-0.5 / std
  float gq = 0.f;
  if (j < L) {
    float d = (float)(j - i);
    gq = coef * expf(-d * d * inv);
  }
  red[j] = (j < L) ? gq : -3.0e38f; __syncthreads();
  for (int s = NT / 2; s > 0; s >>= 1) { if (j < s) red[j] = fmaxf(red[j], red[j + s]); __syncthreads(); }
  const float gmax = red[0];
  if (j < L)
    Wg[((size_t)b * L + i) * L + j] = gq / gmax * wbuf[(size_t)b * L + j];
}

// ---------------------------------------------------------------------------
// Transpose [B,L,D] -> [B,D,L], LDS tiled. block=(32,8), grid=(L/32,D/32,B)
// ---------------------------------------------------------------------------
__global__ void transpose_k(const float* __restrict__ in,
                            float* __restrict__ out, int L, int D) {
  __shared__ float tile[32][33];
  const int b = blockIdx.z;
  const int l0 = blockIdx.x * 32, d0 = blockIdx.y * 32;
  const int tx = threadIdx.x, ty = threadIdx.y;
  for (int r = ty; r < 32; r += 8) {
    int l = l0 + r, d = d0 + tx;
    tile[r][tx] = (l < L && d < D) ? in[((size_t)b * L + l) * D + d] : 0.f;
  }
  __syncthreads();
  for (int r = ty; r < 32; r += 8) {
    int d = d0 + r, l = l0 + tx;
    if (d < D && l < L) out[((size_t)b * D + d) * L + l] = tile[tx][r];
  }
}

// ---------------------------------------------------------------------------
// Residual + concat + max over L:
//  out6[b,off+d]   = max_l(cp[b,l,d] + feats[b,l,d])
//  out6[b,off+D+d] = max_l(cv[b,d,l] + feats[b,l,d])
// ---------------------------------------------------------------------------
__global__ void maxreduce_k(const float* __restrict__ cp,
                            const float* __restrict__ cv,
                            const float* __restrict__ feats,
                            float* __restrict__ out6, int L, int D, int off) {
  const int b = blockIdx.y;
  const int d = blockIdx.x * blockDim.x + threadIdx.x;
  if (d >= D) return;
  float m1 = -3.0e38f, m2 = -3.0e38f;
  for (int l = 0; l < L; ++l) {
    float f = feats[((size_t)b * L + l) * D + d];
    m1 = fmaxf(m1, cp[((size_t)b * L + l) * D + d] + f);
    m2 = fmaxf(m2, cv[((size_t)b * D + d) * L + l] + f);
  }
  out6[(size_t)b * 7680 + off + d]     = m1;
  out6[(size_t)b * 7680 + off + D + d] = m2;
}

// ---------------------------------------------------------------------------
// Host orchestration
// ---------------------------------------------------------------------------
extern "C" void kernel_launch(void* const* d_in, const int* in_sizes, int n_in,
                              void* d_out, int out_size, void* d_ws, size_t ws_size,
                              hipStream_t stream) {
  (void)in_sizes; (void)n_in; (void)out_size; (void)ws_size;
  const int B = 8, D = 1280;

  char* wsp = (char*)d_ws;
  auto carve = [&](size_t bytes) {
    void* p = wsp; wsp += (bytes + 255) & ~(size_t)255; return p;
  };
  float* xT   = (float*)carve((size_t)B * D * 224 * 4);
  float* t1   = (float*)carve((size_t)B * 2 * D * 224 * 4);
  float* t2   = (float*)carve((size_t)B * 2 * D * 224 * 4);
  float* t3   = (float*)carve((size_t)B * D * 224 * 4);
  float* cpb  = (float*)carve((size_t)B * 224 * D * 4);
  float* Wg   = (float*)carve((size_t)B * 224 * 224 * 4);
  float* swb  = (float*)carve((size_t)B * 2 * 224 * 4);
  float* sbuf = (float*)carve((size_t)B * 224 * 4);
  float* wbuf = (float*)carve((size_t)B * 224 * 4);
  float* out6 = (float*)carve((size_t)B * 7680 * 4);
  float* h1   = (float*)carve((size_t)B * 1920 * 4);
  float* h2   = (float*)carve((size_t)B * 1920 * 4);
  float* h3   = (float*)carve((size_t)B * 128 * 4);

  // jax pytree (sorted-dict) leaf order:
  // 0..2 feats; 3..10 cls head; 11..34 hseq(conv12,cp12); 35..58 lseq;
  // 59..82 rbd; 83..90 reg head.
  // branch leaf keys sorted: b1,b2,b3,be1,be2,be3,g1,g2,g3,w1,w2,w3
  enum { kB1 = 0, kB2, kB3, kBE1, kBE2, kBE3, kG1, kG2, kG3, kW1, kW2, kW3 };
  struct Br { int fi, pconv, pcp, L, off; };
  const Br br[3] = {
    {0, 59, 71, 224, 0},     // rbd
    {1, 11, 23, 192, 2560},  // hseq
    {2, 35, 47, 160, 5120},  // lseq
  };
  auto F = [&](int idx) { return (const float*)d_in[idx]; };

  for (int bi = 0; bi < 3; ++bi) {
    const float* feats = F(br[bi].fi);
    const int L = br[bi].L;
    const int pc = br[bi].pconv, pp = br[bi].pcp;

    transpose_k<<<dim3((L + 31) / 32, (D + 31) / 32, B), dim3(32, 8), 0, stream>>>(
        feats, xT, L, D);

    dim3 gBig((L + BN - 1) / BN, (2 * D + BM - 1) / BM, B);

    // ---- context_pool stack (cp params) ----
    conv1d_wmma_k<<<gBig, NT, 0, stream>>>(xT, F(pp + kW1), F(pp + kB1), t1, D, 2 * D, L);
    layernorm2_k<<<B, NT, 0, stream>>>(t1, F(pp + kG1), F(pp + kBE1), 2 * D, L);
    conv1d_wmma_k<<<gBig, NT, 0, stream>>>(t1, F(pp + kW2), F(pp + kB2), t2, 2 * D, 2 * D, L);
    layernorm2_k<<<B, NT, 0, stream>>>(t2, F(pp + kG2), F(pp + kBE2), 2 * D, L);
    conv1d_wmma_k<<<dim3((L + BN - 1) / BN, 1, B), NT, 0, stream>>>(
        t2, F(pp + kW3), F(pp + kB3), swb, 2 * D, 2, L);
    layernorm2_k<<<B, NT, 0, stream>>>(swb, F(pp + kG3), F(pp + kBE3), 2, L);
    softmax_k<<<dim3(2, B), NT, 0, stream>>>(swb, sbuf, wbuf, L);
    gaussw_k<<<dim3(L, B), NT, 0, stream>>>(sbuf, wbuf, Wg, L);
    // cp = Wg @ feats : [L,L] x [L,D]
    gemm_wmma_k<<<dim3((D + BN - 1) / BN, (L + BM - 1) / BM, B), NT, 0, stream>>>(
        Wg, feats, nullptr, cpb, L, D, L,
        (long long)L * L, (long long)L * D, (long long)L * D,
        L, D, D, 0, 0);

    // ---- conv stack (conv params) ----
    conv1d_wmma_k<<<gBig, NT, 0, stream>>>(xT, F(pc + kW1), F(pc + kB1), t1, D, 2 * D, L);
    layernorm2_k<<<B, NT, 0, stream>>>(t1, F(pc + kG1), F(pc + kBE1), 2 * D, L);
    conv1d_wmma_k<<<gBig, NT, 0, stream>>>(t1, F(pc + kW2), F(pc + kB2), t2, 2 * D, 2 * D, L);
    layernorm2_k<<<B, NT, 0, stream>>>(t2, F(pc + kG2), F(pc + kBE2), 2 * D, L);
    conv1d_wmma_k<<<dim3((L + BN - 1) / BN, (D + BM - 1) / BM, B), NT, 0, stream>>>(
        t2, F(pc + kW3), F(pc + kB3), t3, 2 * D, D, L);
    layernorm2_k<<<B, NT, 0, stream>>>(t3, F(pc + kG3), F(pc + kBE3), D, L);

    maxreduce_k<<<dim3((D + NT - 1) / NT, B), NT, 0, stream>>>(
        cpb, t3, feats, out6, L, D, br[bi].off);
  }

  // ---- MLP heads (Dropout / LeakyReLU(slope=1) are identity) ----
  auto head = [&](int base, float* dst, int act) {
    gemm_wmma_k<<<dim3((1920 + BN - 1) / BN, 1, 1), NT, 0, stream>>>(
        out6, F(base + 0), F(base + 1), h1, B, 1920, 7680, 0, 0, 0,
        7680, 7680, 1920, 1, 0);
    gemm_wmma_k<<<dim3((1920 + BN - 1) / BN, 1, 1), NT, 0, stream>>>(
        h1, F(base + 2), F(base + 3), h2, B, 1920, 1920, 0, 0, 0,
        1920, 1920, 1920, 1, 0);
    gemm_wmma_k<<<dim3((120 + BN - 1) / BN, 1, 1), NT, 0, stream>>>(
        h2, F(base + 4), F(base + 5), h3, B, 120, 1920, 0, 0, 0,
        1920, 1920, 120, 1, 0);
    gemm_wmma_k<<<dim3(1, 1, 1), NT, 0, stream>>>(
        h3, F(base + 6), F(base + 7), dst, B, 1, 120, 0, 0, 0,
        120, 120, 1, 1, act);
  };
  head(3,  (float*)d_out,     1);  // cls (sigmoid)
  head(83, (float*)d_out + 8, 0);  // reg
}
